// KPConvSimpleBlock_67714454389197
// MI455X (gfx1250) — matrix operand order repaired and checked by reading, not compiled, not run on hardware
//
#include <hip/hip_runtime.h>

#define NPTS   131072
#define MNB    34
#define KP     15
#define CIN    32
#define COUT   64
#define PINF   0.04f
#define BN_EPS 1e-5f
#define SLOPE  0.2f

typedef __attribute__((ext_vector_type(16))) __bf16 v16bf;
typedef __attribute__((ext_vector_type(8)))  float  v8f;

union FragB16 {
  uint4 q[2];
  unsigned short u[16];
  v16bf v;
};

__device__ __forceinline__ unsigned short f2bf(float f) {
  unsigned int u = __float_as_uint(f);
  u = (u + 0x7FFFu + ((u >> 16) & 1u)) >> 16;   // round-to-nearest-even
  return (unsigned short)u;
}

__device__ __forceinline__ unsigned int f2bf_pk(float lo, float hi) {
  return (unsigned int)f2bf(lo) | ((unsigned int)f2bf(hi) << 16);
}

// 16-bit A-matrix (16x32) K index for (slot = 2j+e, lane half h):
// h=0: V0..V3 -> K 0..7, V4..V7 -> K 16..23 ; h=1: K 8..15, 24..31
__device__ __forceinline__ int kposA(int slot, int h) {
  return (slot < 8) ? (8 * h + slot) : (8 * h + 8 + slot);
}

// ---------------------------------------------------------------------------
// Pack weight [15][32][64] f32 -> bf16 fragments in per-lane WMMA-B order.
// linear t = ((dt*16 + k)*32 + lane)*16 + slot ; B layout: lane half h gives
// K rows 16*h + slot (contiguous), column d = lane%16 + 16*dt.
// ---------------------------------------------------------------------------
__global__ void pack_weight_kernel(const float* __restrict__ weight,
                                   unsigned short* __restrict__ w2pack) {
  int t = blockIdx.x * 256 + threadIdx.x;
  if (t >= 4 * 16 * 32 * 16) return;
  int slot = t & 15;
  int lane = (t >> 4) & 31;
  int k    = (t >> 9) & 15;
  int dt   = t >> 13;
  int h = lane >> 4;
  int c = 16 * h + slot;              // C_IN index (K-dim of step-3 GEMM chunk)
  int d = (lane & 15) + 16 * dt;      // C_OUT index
  float v = (k < KP) ? weight[(k * CIN + c) * COUT + d] : 0.f;
  w2pack[t] = f2bf(v);
}

// ---------------------------------------------------------------------------
// Main fused KPConv kernel: 4 waves (128 threads) per 16-point tile.
// Each wave privately stages+convolves 4 points (step 2), then computes one
// 16-wide C_OUT tile of the [16 x 480] x [480 x 64] GEMM (step 3).
// ---------------------------------------------------------------------------
__global__ __launch_bounds__(128) void kpconv_tile_kernel(
    const float* __restrict__ xyz, const float* __restrict__ feats,
    const int* __restrict__ nidx, const float* __restrict__ kpts,
    const unsigned short* __restrict__ w2pack, float* __restrict__ out) {
  __shared__ __align__(16) unsigned short sWF[16][16][CIN];  // bf16 wf [n][k][c] (shared)
  __shared__ __align__(16) unsigned short sF[4][CIN][72];    // bf16 feats [c][m] per wave
  __shared__ float4 sNBO[4][64];                             // neighbor offsets per wave
  __shared__ float4 sKP[16];

  const int tid  = threadIdx.x;
  const int lane = tid & 31;
  const int wv   = tid >> 5;           // wave id 0..3
  const int col  = lane & 15;
  const int h    = lane >> 4;
  const int n0   = blockIdx.x * 16;

  if (tid < 16) {
    float4 kp = {0.f, 0.f, 0.f, 0.f};
    if (tid < KP) {
      kp.x = kpts[tid * 3 + 0];
      kp.y = kpts[tid * 3 + 1];
      kp.z = kpts[tid * 3 + 2];
    }
    sKP[tid] = kp;
  }
  __syncthreads();
  const float4 kp = sKP[col];          // this lane's kernel point (A row = col)
  const bool kvalid = (col < KP);

  // ---- Step 1+2: per point, influence weights x gathered features --------
  for (int pl = 0; pl < 4; ++pl) {
    const int p = wv * 4 + pl;
    const int n = n0 + p;
    if (pl < 3) __builtin_prefetch(&nidx[(n + 1) * MNB], 0, 3);

    const float px = xyz[n * 3 + 0];
    const float py = xyz[n * 3 + 1];
    const float pz = xyz[n * 3 + 2];

    // neighbor ids for this point live in two VGPRs (m = lane, m = 32+lane)
    const int idA = nidx[n * MNB + lane];                         // m = 0..31
    const int idB = (lane < MNB - 32) ? nidx[n * MNB + 32 + lane] : -1;

    {
      float4 o = {1.0e6f, 1.0e6f, 1.0e6f, 0.f};
      if ((unsigned)idA < (unsigned)NPTS) {
        o.x = xyz[idA * 3 + 0] - px;
        o.y = xyz[idA * 3 + 1] - py;
        o.z = xyz[idA * 3 + 2] - pz;
      }
      sNBO[wv][lane] = o;
      float4 o2 = {1.0e6f, 1.0e6f, 1.0e6f, 0.f};
      if ((unsigned)idB < (unsigned)NPTS) {
        o2.x = xyz[idB * 3 + 0] - px;
        o2.y = xyz[idB * 3 + 1] - py;
        o2.z = xyz[idB * 3 + 2] - pz;
      }
      sNBO[wv][lane + 32] = o2;
    }
    __builtin_amdgcn_wave_barrier();

    // gather features: lane = channel; broadcast ids via compile-time shfl,
    // keep ~16 loads in flight, pack bf16 pairs into b32 LDS stores
    #pragma unroll
    for (int mb = 0; mb < 64; mb += 16) {
      float v[16];
      #pragma unroll
      for (int i = 0; i < 16; ++i) {
        const int m = mb + i;
        if (m < 32) {
          v[i] = feats[__shfl(idA, m) * CIN + lane];
        } else if (m < MNB) {
          v[i] = feats[__shfl(idB, m - 32) * CIN + lane];
        } else {
          v[i] = 0.f;
        }
      }
      #pragma unroll
      for (int i = 0; i < 16; i += 2) {
        *(unsigned int*)&sF[wv][lane][mb + i] = f2bf_pk(v[i], v[i + 1]);
      }
    }
    __builtin_amdgcn_wave_barrier();

    v8f acc0 = {0, 0, 0, 0, 0, 0, 0, 0};
    v8f acc1 = {0, 0, 0, 0, 0, 0, 0, 0};
    #pragma unroll
    for (int chunk = 0; chunk < 2; ++chunk) {
      // A = w[k=16 rows, m=32] computed in ISA A-layout
      FragB16 a;
      #pragma unroll
      for (int slot = 0; slot < 16; ++slot) {
        int m = kposA(slot, h) + 32 * chunk;
        float w = 0.f;
        if (kvalid && m < MNB) {
          float dx = sNBO[wv][m].x - kp.x;
          float dy = sNBO[wv][m].y - kp.y;
          float dz = sNBO[wv][m].z - kp.z;
          float dist = sqrtf(dx * dx + dy * dy + dz * dz);
          w = fmaxf(0.f, 1.f - dist * (1.f / PINF));
        }
        a.u[slot] = f2bf(w);
      }
      // B = feats[m=32 chunk, c=16]; B layout: lane col = c, K rows 8h+slot, 16+8h+slot-8
      const int mb = 8 * h + 32 * chunk;
      FragB16 b0, b1;
      b0.q[0] = *(const uint4*)&sF[wv][col][mb];
      b0.q[1] = *(const uint4*)&sF[wv][col][mb + 16];
      b1.q[0] = *(const uint4*)&sF[wv][col + 16][mb];
      b1.q[1] = *(const uint4*)&sF[wv][col + 16][mb + 16];
      acc0 = __builtin_amdgcn_wmma_f32_16x16x32_bf16(false, a.v, false, b0.v,
                                                     (short)0, acc0, false, false);
      acc1 = __builtin_amdgcn_wmma_f32_16x16x32_bf16(false, a.v, false, b1.v,
                                                     (short)0, acc1, false, false);
    }
    // D[k=16, c=16] per ctile: lane col = c, VGPR r -> k = r + 8h
    #pragma unroll
    for (int r = 0; r < 8; ++r) {
      int k = r + 8 * h;
      sWF[p][k][col]      = f2bf(acc0[r]);
      sWF[p][k][col + 16] = f2bf(acc1[r]);
    }
    __builtin_amdgcn_wave_barrier();
  }
  __syncthreads();

  // ---- Step 3: out[16 pts, 64] = wf[16, 480] x W2[480, 64]; dt = wave ----
  {
    const int dt = wv;
    v8f acc = {0, 0, 0, 0, 0, 0, 0, 0};
    for (int k = 0; k < 16; ++k) {
      FragB16 a, b;
      // A row = point (col), kc chunk = k; A-layout K slots are contiguous
      // c = 16h..16h+15 (matches packed B)
      a.q[0] = *(const uint4*)&sWF[col][k][16 * h];
      a.q[1] = *(const uint4*)&sWF[col][k][16 * h + 8];
      const uint4* pb =
          (const uint4*)&w2pack[(((dt * 16 + k) * 32) + lane) * 16];
      b.q[0] = pb[0];
      b.q[1] = pb[1];
      acc = __builtin_amdgcn_wmma_f32_16x16x32_bf16(false, a.v, false, b.v,
                                                    (short)0, acc, false, false);
    }
    #pragma unroll
    for (int r = 0; r < 8; ++r) {
      out[(n0 + r + 8 * h) * COUT + col + 16 * dt] = acc[r];
    }
  }
}

// ---------------------------------------------------------------------------
// BatchNorm statistics: one block per output channel, deterministic tree sum.
// ---------------------------------------------------------------------------
__global__ void bn_stats_kernel(const float* __restrict__ out,
                                float* __restrict__ stats) {
  __shared__ float rs[256], rs2[256];
  const int d = blockIdx.x;
  const int t = threadIdx.x;
  float s = 0.f, s2 = 0.f;
  for (int n = t; n < NPTS; n += 256) {
    float v = out[n * COUT + d];
    s += v;
    s2 += v * v;
  }
  rs[t] = s;
  rs2[t] = s2;
  __syncthreads();
  for (int off = 128; off > 0; off >>= 1) {
    if (t < off) {
      rs[t] += rs[t + off];
      rs2[t] += rs2[t + off];
    }
    __syncthreads();
  }
  if (t == 0) {
    float mean = rs[0] * (1.f / (float)NPTS);
    float var = rs2[0] * (1.f / (float)NPTS) - mean * mean;
    stats[d] = mean;
    stats[COUT + d] = rsqrtf(var + BN_EPS);
  }
}

// ---------------------------------------------------------------------------
// Apply BN + LeakyReLU in place.
// ---------------------------------------------------------------------------
__global__ void bn_apply_kernel(float* __restrict__ out,
                                const float* __restrict__ stats,
                                const float* __restrict__ gamma,
                                const float* __restrict__ beta) {
  int i = blockIdx.x * 256 + threadIdx.x;
  if (i >= NPTS * COUT) return;
  int d = i & (COUT - 1);
  float x = out[i];
  x = (x - stats[d]) * stats[COUT + d] * gamma[d] + beta[d];
  out[i] = (x >= 0.f) ? x : SLOPE * x;
}

extern "C" void kernel_launch(void* const* d_in, const int* in_sizes, int n_in,
                              void* d_out, int out_size, void* d_ws, size_t ws_size,
                              hipStream_t stream) {
  (void)in_sizes; (void)n_in; (void)out_size; (void)ws_size;
  const float* xyz    = (const float*)d_in[0];
  const float* feats  = (const float*)d_in[1];
  const int*   nidx   = (const int*)d_in[2];
  const float* weight = (const float*)d_in[3];
  const float* kpts   = (const float*)d_in[4];
  const float* gamma  = (const float*)d_in[5];
  const float* beta   = (const float*)d_in[6];
  float* out = (float*)d_out;

  unsigned short* w2pack = (unsigned short*)d_ws;                // 64 KB
  float* stats = (float*)((char*)d_ws + 64 * 1024);              // 512 B

  pack_weight_kernel<<<128, 256, 0, stream>>>(weight, w2pack);
  kpconv_tile_kernel<<<NPTS / 16, 128, 0, stream>>>(xyz, feats, nidx, kpts,
                                                    w2pack, out);
  bn_stats_kernel<<<COUT, 256, 0, stream>>>(out, stats);
  bn_apply_kernel<<<(NPTS * COUT + 255) / 256, 256, 0, stream>>>(out, stats,
                                                                 gamma, beta);
}
